// GIN_7095285973355
// MI455X (gfx1250) — compile-verified, matrix operand order checked by reading, not run on hardware
//
#include <hip/hip_runtime.h>
#include <hip/hip_bf16.h>

typedef __attribute__((ext_vector_type(16))) _Float16 v16h;
typedef __attribute__((ext_vector_type(8)))  float    v8f;
typedef unsigned int u32x4 __attribute__((ext_vector_type(4)));
typedef int          i32x8 __attribute__((ext_vector_type(8)));
typedef int          i32x4 __attribute__((ext_vector_type(4)));

#define LEAKY_SLOPE 0.2f
#define BN_EPS_F 1e-5f

// ---------------------------------------------------------------------------
// Tensor Data Mover descriptors (CDNA5 ISA):
// group0: [1:0]count=1, [63:32]lds_addr, [120:64]global_addr, [127:126]type=2
// group1: [17:16]data_size(1=2B), tensor_dim0/1 = huge (no OOB), tile dims,
//         [207:160] tensor_dim0_stride (element units).
// 2D tile: tile_w elements per line, tile_h lines, row stride in elements.
// This toolchain exposes the 6-arg builtin: (g0, g1, g2, g3, g4, cpol).
// ---------------------------------------------------------------------------
static __device__ __forceinline__ void tdm_desc(u32x4& g0, i32x8& g1,
                                                unsigned int lds_addr, const void* gptr,
                                                unsigned int tile_w, unsigned int tile_h,
                                                unsigned long long stride_elems) {
  unsigned long long ga = (unsigned long long)(uintptr_t)gptr & 0x01FFFFFFFFFFFFFFull;
  g0[0] = 1u;                                   // count=1 (valid user descriptor)
  g0[1] = lds_addr;
  g0[2] = (unsigned int)ga;
  g0[3] = (unsigned int)(ga >> 32) | 0x80000000u;  // type=2 ("image")
  g1[0] = (int)(1u << 16);                      // data_size = 2 bytes
  g1[1] = (int)0xFFFF0000u;                     // tensor_dim0 = 0x7FFFFFFF (lo16<<16)
  g1[2] = (int)0xFFFF7FFFu;                     // td0 hi16 | tensor_dim1 lo16
  g1[3] = (int)(0x7FFFu | (tile_w << 16));      // td1 hi16 | tile_dim0
  g1[4] = (int)(tile_h & 0xFFFFu);              // tile_dim1 (tile_dim2 = 0)
  g1[5] = (int)(unsigned int)stride_elems;      // tensor_dim0_stride lo32
  g1[6] = (int)((stride_elems >> 32) & 0xFFFFull);
  g1[7] = 0;
}

static __device__ __forceinline__ void tdm_load_2d(unsigned int lds_addr, const void* gptr,
                                                   unsigned int tile_w, unsigned int tile_h,
                                                   unsigned long long stride_elems) {
  u32x4 g0; i32x8 g1; i32x4 z4 = {0, 0, 0, 0}; i32x8 z8 = {0, 0, 0, 0, 0, 0, 0, 0};
  tdm_desc(g0, g1, lds_addr, gptr, tile_w, tile_h, stride_elems);
  __builtin_amdgcn_tensor_load_to_lds(g0, g1, z4, z4, z8, 0);
}

static __device__ __forceinline__ void tdm_store_2d(unsigned int lds_addr, void* gptr,
                                                    unsigned int tile_w, unsigned int tile_h,
                                                    unsigned long long stride_elems) {
  u32x4 g0; i32x8 g1; i32x4 z4 = {0, 0, 0, 0}; i32x8 z8 = {0, 0, 0, 0, 0, 0, 0, 0};
  tdm_desc(g0, g1, lds_addr, gptr, tile_w, tile_h, stride_elems);
  __builtin_amdgcn_tensor_store_from_lds(g0, g1, z4, z4, z8, 0);
}

// ---------------------------------------------------------------------------
// WMMA fragment loaders (CDNA5 ISA 16-bit layouts, CDNA5 ISA):
//   A (16x32 MxK): lane L row M=L%16; element e -> K = k0+(e%8)+16*(e/8)+8*(L/16)
//   B (32x16 KxN): pre-swizzled so each lane's 16 halves are contiguous.
// ---------------------------------------------------------------------------
static __device__ __forceinline__ v16h frag_a(const _Float16* __restrict__ s,
                                              int ld, int row, int k0, int lane) {
  const _Float16* p = s + (size_t)row * ld + k0 + ((lane >> 4) << 3);
  v16h a;
#pragma unroll
  for (int e = 0; e < 8; ++e) a[e] = p[e];
#pragma unroll
  for (int e = 0; e < 8; ++e) a[8 + e] = p[16 + e];
  return a;
}

static __device__ __forceinline__ v16h frag_b_swz(const _Float16* __restrict__ s,
                                                  int grp, int n, int NSN, int lane) {
  const _Float16* p = s + (((size_t)(grp * NSN + n)) << 9) + (lane << 4);
  v16h b;
#pragma unroll
  for (int e = 0; e < 16; ++e) b[e] = p[e];
  return b;
}

union HU8 { uint4 u; _Float16 h[8]; };

// ---------------------------------------------------------------------------
// Tiled WMMA GEMM: out = act(X @ Wsw + bias); optional column-stats emission.
// X: M x K f16 (M%128==0), Wsw: fragment-swizzled, bias: N f32.
// Tiles staged via the Tensor Data Mover; output tile stored via TDM.
// Block: 256 threads / 8 wave32 waves; tile 128 x N; K staged in 64-chunks.
// ---------------------------------------------------------------------------
template <int K, int N>
__global__ __launch_bounds__(256) void k_gemm(const _Float16* __restrict__ X,
                                              const _Float16* __restrict__ Wsw,
                                              const float* __restrict__ bias,
                                              void* __restrict__ out,
                                              float* __restrict__ stats,
                                              int leaky, int out_f32, int emit_stats) {
  constexpr int NSN = N / 16;
  __shared__ __attribute__((aligned(16))) _Float16 sX[128 * 64];
  __shared__ __attribute__((aligned(16))) _Float16 sW[64 * N];
  __shared__ __attribute__((aligned(16))) _Float16 sOut[128 * N];
  const int tid  = threadIdx.x;
  const int lane = tid & 31;
  const int wave = tid >> 5;
  const int blkRow = blockIdx.x * 128;

  v8f acc[NSN];
  v8f zero8 = {};
#pragma unroll
  for (int t = 0; t < NSN; ++t) acc[t] = zero8;

#pragma unroll 1
  for (int kt = 0; kt < K; kt += 64) {
    __syncthreads();                    // previous tile fully consumed
    if (tid < 32) {                     // wave 0 drives the Tensor Data Mover
      tdm_load_2d((unsigned int)(uintptr_t)sX,
                  X + (size_t)blkRow * K + kt, 64u, 128u, (unsigned long long)K);
      tdm_load_2d((unsigned int)(uintptr_t)sW,
                  Wsw + ((size_t)(kt >> 5) * NSN << 9),
                  (unsigned int)(16 * N), 4u, (unsigned long long)(16 * N));
      __builtin_amdgcn_s_wait_tensorcnt(0);
    }
    __syncthreads();

#pragma unroll
    for (int k0 = 0; k0 < 64; k0 += 32) {
      v16h a = frag_a(sX, 64, (wave << 4) + (lane & 15), k0, lane);
#pragma unroll
      for (int t = 0; t < NSN; ++t) {
        v16h b = frag_b_swz(sW, k0 >> 5, t, NSN, lane);
        acc[t] = __builtin_amdgcn_wmma_f32_16x16x32_f16(
            false, a, false, b, (short)0, acc[t], false, false);
      }
    }
  }

  // ---- epilogue
  const int half = lane >> 4;
  const int cl   = lane & 15;
  if (out_f32) {
#pragma unroll
    for (int t = 0; t < NSN; ++t) {
      int col = t * 16 + cl;
      float bv = bias[col];
#pragma unroll
      for (int r = 0; r < 8; ++r) {
        int grow = blkRow + (wave << 4) + r + (half << 3);
        float v = acc[t][r] + bv;
        if (leaky) v = (v >= 0.0f) ? v : LEAKY_SLOPE * v;
        reinterpret_cast<float*>(out)[(size_t)grow * N + col] = v;
      }
    }
    return;
  }
#pragma unroll
  for (int t = 0; t < NSN; ++t) {
    int col = t * 16 + cl;
    float bv = bias[col];
#pragma unroll
    for (int r = 0; r < 8; ++r) {
      int row = (wave << 4) + r + (half << 3);
      float v = acc[t][r] + bv;
      if (leaky) v = (v >= 0.0f) ? v : LEAKY_SLOPE * v;
      sOut[row * N + col] = (_Float16)v;
    }
  }
  __syncthreads();
  if (emit_stats) {
    for (int col = tid; col < N; col += 256) {
      float s = 0.0f, ss = 0.0f;
#pragma unroll 4
      for (int r = 0; r < 128; ++r) {
        float v = (float)sOut[r * N + col];
        s += v; ss += v * v;
      }
      atomicAdd(&stats[col], s);
      atomicAdd(&stats[N + col], ss);
    }
  }
  if (tid < 32)  // contiguous 128*N halves: TDM store; ENDPGM waits for idle
    tdm_store_2d((unsigned int)(uintptr_t)sOut,
                 reinterpret_cast<_Float16*>(out) + (size_t)blkRow * N,
                 (unsigned int)(16 * N), 8u, (unsigned long long)(16 * N));
}

// ---------------------------------------------------------------------------
// Per-graph GIN aggregation via WMMA: Y = (A^T + I) @ X, one block per graph.
// S padded to 64x64 f16; X (50 x DX) padded to 64 rows, swizzled into
// B-fragment layout during staging; 50 valid output rows stored via TDM.
// ---------------------------------------------------------------------------
template <int DX>
__global__ __launch_bounds__(256) void k_aggregate(const float* __restrict__ A,
                                                   const _Float16* __restrict__ X,
                                                   _Float16* __restrict__ Y) {
  constexpr int NSN = DX / 16;
  __shared__ __attribute__((aligned(16))) _Float16 sS[64 * 64];
  __shared__ __attribute__((aligned(16))) _Float16 sXs[64 * DX];   // swizzled
  __shared__ __attribute__((aligned(16))) _Float16 sOut[64 * DX];
  const int g    = blockIdx.x;
  const int tid  = threadIdx.x;
  const int lane = tid & 31;
  const int wave = tid >> 5;
  const float* Ag = A + (size_t)g * 2500;

#pragma unroll
  for (int j = 0; j < 16; ++j) sS[tid + j * 256] = (_Float16)0.0f;
  __syncthreads();
  for (int idx = tid; idx < 2500; idx += 256) {
    int i = idx / 50, jj = idx % 50;
    sS[jj * 64 + i] = (_Float16)Ag[idx];       // S[j][i] = A[i][j]
  }
  __syncthreads();
  if (tid < 50) sS[tid * 64 + tid] = sS[tid * 64 + tid] + (_Float16)1.0f;  // +I

  // stage X with B-fragment swizzle: k = node row, c = feature col
  constexpr int U = 64 * DX / 8;
#pragma unroll
  for (int j = 0; j < U / 256; ++j) {
    int idx = tid + j * 256;
    int k = idx / (DX / 8), seg = idx % (DX / 8);
    HU8 v; v.u = make_uint4(0u, 0u, 0u, 0u);
    if (k < 50)
      v.u = *reinterpret_cast<const uint4*>(X + ((size_t)g * 50 + k) * DX + seg * 8);
    int grp = k >> 5;
    int kk  = k & 31;
    int hlf = (kk >> 3) & 1;
    int e   = (kk & 7) | (((kk >> 4) & 1) << 3);
#pragma unroll
    for (int q = 0; q < 8; ++q) {
      int c = seg * 8 + q;
      int ln = (c & 15) | (hlf << 4);
      sXs[(((grp * NSN) + (c >> 4)) << 9) + (ln << 4) + e] = v.h[q];
    }
  }
  __syncthreads();

  constexpr int TOT = 4 * NSN;
  for (int t = wave; t < TOT; t += 8) {
    int m = t / NSN, n = t % NSN;
    v8f acc = {};
#pragma unroll
    for (int k0 = 0; k0 < 64; k0 += 32) {
      v16h a = frag_a(sS, 64, (m << 4) + (lane & 15), k0, lane);
      v16h b = frag_b_swz(sXs, k0 >> 5, n, NSN, lane);
      acc = __builtin_amdgcn_wmma_f32_16x16x32_f16(
          false, a, false, b, (short)0, acc, false, false);
    }
    const int cl = lane & 15, hlf = lane >> 4;
#pragma unroll
    for (int r = 0; r < 8; ++r)
      sOut[((m << 4) + r + (hlf << 3)) * DX + (n << 4) + cl] = (_Float16)acc[r];
  }
  __syncthreads();
  if (tid < 32)  // 50 valid rows are contiguous in Y: one TDM store
    tdm_store_2d((unsigned int)(uintptr_t)sOut, Y + (size_t)g * 50 * DX,
                 (unsigned int)DX, 50u, (unsigned long long)DX);
}

// ---------------------------------------------------------------------------
// Helpers
// ---------------------------------------------------------------------------
// f32 -> f16 B-fragment-swizzled weights, K padded, optional per-row scale
__global__ void k_cvt_w_swz(const float* __restrict__ W, _Float16* __restrict__ Wh,
                            const float* __restrict__ scsh, int din, int N, int total) {
  int idx = blockIdx.x * blockDim.x + threadIdx.x;
  if (idx >= total) return;
  int e    = idx & 15;
  int lane = (idx >> 4) & 31;
  int grp  = idx >> 9;
  int NSN  = N >> 4;
  int kt32 = grp / NSN, n = grp % NSN;
  int k = (kt32 << 5) + (e & 7) + ((e >> 3) << 4) + (((lane >> 4) & 1) << 3);
  int c = (n << 4) + (lane & 15);
  float v = (k < din) ? W[k * N + c] : 0.0f;
  if (scsh) v *= scsh[k];
  Wh[idx] = (_Float16)v;
}

// bias' = b + sh @ W  (fold BN shift through the next linear layer)
__global__ void k_foldbias(const float* __restrict__ b, const float* __restrict__ W,
                           const float* __restrict__ scsh, float* __restrict__ bout,
                           int N) {
  int c = threadIdx.x;
  if (c >= N) return;
  float s = b[c];
  for (int k = 0; k < 256; ++k) s += scsh[256 + k] * W[k * N + c];
  bout[c] = s;
}

__global__ void k_cvt_x0(const float* __restrict__ A, _Float16* __restrict__ X0) {
  size_t idx = (size_t)blockIdx.x * blockDim.x + threadIdx.x;  // 409600*64
  size_t row = idx >> 6;
  int col = (int)(idx & 63);
  X0[idx] = (_Float16)((col < 50) ? A[row * 50 + col] : 0.0f);
}

__global__ void k_zero(float* p, int n) {
  int i = blockIdx.x * blockDim.x + threadIdx.x;
  if (i < n) p[i] = 0.0f;
}

__global__ __launch_bounds__(256) void k_colstats(const _Float16* __restrict__ X,
                                                  float* __restrict__ stats,
                                                  int M, int rowsPerBlk) {
  int c = threadIdx.x;
  int r0 = blockIdx.x * rowsPerBlk;
  int r1 = r0 + rowsPerBlk; if (r1 > M) r1 = M;
  float s = 0.0f, ss = 0.0f;
  for (int r = r0; r < r1; ++r) {
    float v = (float)X[(size_t)r * 256 + c];
    s += v; ss += v * v;
  }
  atomicAdd(&stats[c], s);
  atomicAdd(&stats[256 + c], ss);
}

__global__ void k_prepbn(const float* __restrict__ stats,
                         const float* __restrict__ gamma,
                         const float* __restrict__ beta,
                         float* __restrict__ scsh, float invcnt) {
  int c = threadIdx.x;  // 256
  float mu  = stats[c] * invcnt;
  float var = stats[256 + c] * invcnt - mu * mu;
  float sc  = gamma[c] * rsqrtf(var + BN_EPS_F);
  scsh[c] = sc;
  scsh[256 + c] = beta[c] - mu * sc;
}

__global__ __launch_bounds__(256) void k_pool(const _Float16* __restrict__ X,
                                              _Float16* __restrict__ P) {
  int g = blockIdx.x, c = threadIdx.x;
  float s = 0.0f;
  for (int r = 0; r < 50; ++r)
    s += (float)X[((size_t)g * 50 + r) * 256 + c];
  P[(size_t)g * 256 + c] = (_Float16)s;
}

// ---------------------------------------------------------------------------
// Host orchestration
// ---------------------------------------------------------------------------
extern "C" void kernel_launch(void* const* d_in, const int* in_sizes, int n_in,
                              void* d_out, int out_size, void* d_ws, size_t ws_size,
                              hipStream_t stream) {
  (void)in_sizes; (void)n_in; (void)out_size; (void)ws_size;
  const float* A = (const float*)d_in[0];
  const float* W1[3] = {(const float*)d_in[1],  (const float*)d_in[7],  (const float*)d_in[13]};
  const float* b1[3] = {(const float*)d_in[2],  (const float*)d_in[8],  (const float*)d_in[14]};
  const float* gm[3] = {(const float*)d_in[3],  (const float*)d_in[9],  (const float*)d_in[15]};
  const float* bt[3] = {(const float*)d_in[4],  (const float*)d_in[10], (const float*)d_in[16]};
  const float* W2[3] = {(const float*)d_in[5],  (const float*)d_in[11], (const float*)d_in[17]};
  const float* b2[3] = {(const float*)d_in[6],  (const float*)d_in[12], (const float*)d_in[18]};
  const float* bn_g = (const float*)d_in[19];
  const float* bn_b = (const float*)d_in[20];
  const float* fcW  = (const float*)d_in[21];
  const float* fcb  = (const float*)d_in[22];

  const size_t MROWS = (size_t)8192 * 50;             // 409600
  const size_t SZ = MROWS * 256 * sizeof(_Float16);   // 200 MiB per buffer
  char* ws = (char*)d_ws;
  _Float16* bufA   = (_Float16*)(ws);
  _Float16* bufB   = (_Float16*)(ws + SZ);
  _Float16* pooled = (_Float16*)(ws + 2 * SZ);
  float*    stats  = (float*)(ws + 2 * SZ + (size_t)8192 * 256 * sizeof(_Float16));
  float*    scsh   = stats + 512;
  float*    biasbuf= scsh + 512;
  _Float16* warena = (_Float16*)((char*)(biasbuf + 256) + 1024);

  _Float16* W1h[3];
  W1h[0] = warena;                     // 64*256 (K padded 50->64)
  W1h[1] = W1h[0] + 64 * 256;
  W1h[2] = W1h[1] + 256 * 256;
  _Float16* W2dyn = W1h[2] + 256 * 256;   // per-layer BN-scaled W2 (256*256)
  _Float16* fcdyn = W2dyn + 256 * 256;    // BN-scaled FC weight (256*64)

  // static W1 conversions (no BN dependence)
  k_cvt_w_swz<<<(64 * 256 + 255) / 256, 256, 0, stream>>>(W1[0], W1h[0], nullptr, 50, 256, 64 * 256);
  k_cvt_w_swz<<<(256 * 256 + 255) / 256, 256, 0, stream>>>(W1[1], W1h[1], nullptr, 256, 256, 256 * 256);
  k_cvt_w_swz<<<(256 * 256 + 255) / 256, 256, 0, stream>>>(W1[2], W1h[2], nullptr, 256, 256, 256 * 256);

  // x0 = adjacency rows, f16, K padded 50 -> 64
  k_cvt_x0<<<(unsigned)(MROWS * 64 / 256), 256, 0, stream>>>(A, bufA);

  const float invM = 1.0f / (float)MROWS;
  _Float16* X  = bufA;
  _Float16* Yb = bufB;
  for (int l = 0; l < 3; ++l) {
    // Yb = (A^T + I) @ X
    if (l == 0) k_aggregate<64><<<8192, 256, 0, stream>>>(A, X, Yb);
    else        k_aggregate<256><<<8192, 256, 0, stream>>>(A, X, Yb);
    // X = leaky(Yb @ W1 + b1), BN column stats emitted from epilogue
    k_zero<<<2, 256, 0, stream>>>(stats, 512);
    if (l == 0)
      k_gemm<64, 256><<<3200, 256, 0, stream>>>(Yb, W1h[0], b1[0], X, stats, 1, 0, 1);
    else
      k_gemm<256, 256><<<3200, 256, 0, stream>>>(Yb, W1h[l], b1[l], X, stats, 1, 0, 1);
    // fold BN into W2/b2:  bn(x)@W2 + b2 = x@(sc*W2) + (sh@W2 + b2)
    k_prepbn<<<1, 256, 0, stream>>>(stats, gm[l], bt[l], scsh, invM);
    k_cvt_w_swz<<<(256 * 256 + 255) / 256, 256, 0, stream>>>(W2[l], W2dyn, scsh, 256, 256, 256 * 256);
    k_foldbias<<<1, 256, 0, stream>>>(b2[l], W2[l], scsh, biasbuf, 256);
    // Yb = leaky(x' @ W2' + b2')
    k_gemm<256, 256><<<3200, 256, 0, stream>>>(X, W2dyn, biasbuf, Yb, stats, 1, 0, 0);
    _Float16* tmp = X; X = Yb; Yb = tmp;
  }

  // pool + BN folded into final FC
  k_pool<<<8192, 256, 0, stream>>>(X, pooled);
  k_zero<<<2, 256, 0, stream>>>(stats, 512);
  k_colstats<<<64, 256, 0, stream>>>(pooled, stats, 8192, 128);
  k_prepbn<<<1, 256, 0, stream>>>(stats, bn_g, bn_b, scsh, 1.0f / 8192.0f);
  k_cvt_w_swz<<<(256 * 64 + 255) / 256, 256, 0, stream>>>(fcW, fcdyn, scsh, 256, 64, 256 * 64);
  k_foldbias<<<1, 256, 0, stream>>>(fcb, fcW, scsh, biasbuf, 64);
  k_gemm<256, 64><<<64, 256, 0, stream>>>(pooled, fcdyn, biasbuf, d_out, stats, 0, 1, 0);
}
